// PAMNet_5909875000240
// MI455X (gfx1250) — matrix operand order repaired and checked by reading, not compiled, not run on hardware
//
#include <hip/hip_runtime.h>
#include <hip/hip_bf16.h>
#include <math.h>

// ---------------------------------------------------------------------------
// Types for CDNA5 WMMA
// ---------------------------------------------------------------------------
typedef __attribute__((ext_vector_type(16))) _Float16 v16h;
typedef __attribute__((ext_vector_type(8)))  float    v8f;

#define ACT_NONE 0
#define ACT_SILU 1
#define EPI_STORE 0
#define EPI_ADD   1
#define EPI_MUL   2

#define DIMF 128
#define NSPH 7
#define NRAD 6
#define CUT_L 2.4f
#define CUT_G 5.0f
#define PI_F 3.14159265358979323846f

#define KGROUP 4        // k-steps staged per LDS round (4 * 8KB = 32KB)
#define GEMM_WAVES 8    // waves per block -> 128 rows per block

// Spherical Bessel zeros j_l, l=0..6, n=0..5 (host constants in the reference)
__device__ const float SPH_Z[NSPH * NRAD] = {
    3.14159265f,  6.28318531f,  9.42477796f, 12.56637061f, 15.70796327f, 18.84955592f,
    4.49340946f,  7.72525184f, 10.90412166f, 14.06619391f, 17.22075527f, 20.37130296f,
    5.76345920f,  9.09501133f, 12.32294097f, 15.51460301f, 18.68903640f, 21.85387422f,
    6.98793200f, 10.41711855f, 13.69802315f, 16.92362129f, 20.12180617f, 23.30424947f,
    8.18256145f, 11.70490715f, 15.03966471f, 18.30125596f, 21.52541773f, 24.72756568f,
    9.35581211f, 12.96653017f, 16.35470964f, 19.65315286f, 22.90455064f, 26.12775463f,
   10.51283540f, 14.20739209f, 17.64797495f, 20.98346482f, 24.26276805f, 27.50786839f
};

// ---------------------------------------------------------------------------
// Device math helpers (mirror the reference)
// ---------------------------------------------------------------------------
__device__ __forceinline__ float envelope_f(float x) {
    // p = PEXP+1 = 6: 1/x - 28 x^5 + 48 x^6 - 21 x^7
    float x2 = x * x;
    float x4 = x2 * x2;
    float x5 = x4 * x;
    return 1.0f / x - 28.0f * x5 + 48.0f * x5 * x - 21.0f * x5 * x2;
}

__device__ float sph_jn_f(float x, int l) {
    float xs = fmaxf(x, 1e-6f);
    float s = __sinf(xs), c = __cosf(xs);
    float j0 = s / xs;
    float rec;
    if (l == 0) {
        rec = j0;
    } else {
        float jm = j0;
        float jc = s / (xs * xs) - c / xs;
        for (int ll = 1; ll < l; ++ll) {
            float t = (2.0f * ll + 1.0f) / xs * jc - jm;
            jm = jc; jc = t;
        }
        rec = jc;
    }
    float dfac = 1.0f;
    for (int k = 0; k <= l; ++k) dfac *= (2.0f * k + 1.0f);
    float xl = 1.0f;
    for (int k = 0; k < l; ++k) xl *= xs;
    float x2 = xs * xs;
    float ser = xl / dfac *
        (1.0f - x2 / (2.0f * (2.0f * l + 3.0f)) +
         x2 * x2 / (8.0f * (2.0f * l + 3.0f) * (2.0f * l + 5.0f)));
    return (xs < 0.5f + 0.5f * l) ? ser : rec;
}

__device__ __forceinline__ float silu_f(float v) {
    return v / (1.0f + __expf(-v));
}

// ---------------------------------------------------------------------------
// Weight fragment packing: f32 W[K][128] -> f16 B-fragment layout.
// Layout: frag[(((ks*8 + nt)*32) + lane)*16 + e], where
//   col n = nt*16 + (lane&15),  k = ks*32 + (lane>>4)*16 + e
// ---------------------------------------------------------------------------
__global__ void pamnet_pack_w(const float* __restrict__ W, int K,
                              _Float16* __restrict__ frag, int total) {
    int id = blockIdx.x * blockDim.x + threadIdx.x;
    if (id >= total) return;
    int e    = id & 15;
    int lane = (id >> 4) & 31;
    int nt   = (id >> 9) & 7;
    int ks   = id >> 12;
    int n = nt * 16 + (lane & 15);
    int k = ks * 32 + ((lane >> 4) << 4) + e;
    frag[id] = (k < K) ? (_Float16)W[(size_t)k * DIMF + n] : (_Float16)0.0f;
}

// ---------------------------------------------------------------------------
// Generic WMMA GEMM: C[M x 128] = epi( act( A[M x K] @ W + bias ) )
// 8 waves per block; each wave owns a 16-row x 128-col tile (128 rows/block).
// Packed B fragments are staged through LDS in 32KB chunks so all 8 waves
// share one L2 read, and the per-WMMA dependency wait is a short DScnt wait.
// concat3: A row = [ GA[idxA[r]] (128) | GB[idxB[r]] (128) | GC[r] (128) ]
// The three gather row pointers are hoisted out of the k-loop (indices are
// loaded exactly once); per k-step segment choice is a pointer select.
// ---------------------------------------------------------------------------
__global__ __launch_bounds__(256) void pamnet_gemm(
    const float* __restrict__ A, int K, int lda,
    const float* __restrict__ GA, const float* __restrict__ GB,
    const float* __restrict__ GC,
    const int* __restrict__ idxA, const int* __restrict__ idxB, int concat3,
    const _Float16* __restrict__ frag,
    const float* __restrict__ bias,
    float* __restrict__ C, int M, int act, int epi) {

    __shared__ __align__(32) _Float16 sfrag[KGROUP * 4096];

    int wave = threadIdx.x >> 5;
    int lane = threadIdx.x & 31;
    int tile = blockIdx.x * GEMM_WAVES + wave;
    int row0 = tile * 16;
    bool active = (row0 < M);

    int mlane = lane & 15;
    int khalf = lane >> 4;
    int r  = row0 + mlane;
    bool rv = active && (r < M);
    int rr = rv ? r : 0;

    int nk = (K + 31) >> 5;
    const bool kvec = ((K & 31) == 0);  // A rows 32B-aligned groups, no K guard needed

    // Hoisted A-row pointers (gather indices read once, not per k-step).
    const float* p0;
    const float* p1 = nullptr;
    const float* p2 = nullptr;
    if (concat3) {
        p0 = GA + (size_t)idxA[rr] * DIMF;
        p1 = GB + (size_t)idxB[rr] * DIMF;
        p2 = GC + (size_t)rr * DIMF;
    } else {
        p0 = A + (size_t)rr * lda;
    }

    v8f acc[8];
#pragma unroll
    for (int t = 0; t < 8; ++t)
#pragma unroll
        for (int v = 0; v < 8; ++v) acc[t][v] = 0.0f;

    const _Float16 hz = (_Float16)0.0f;

    for (int kg = 0; kg < nk; kg += KGROUP) {
        int ngroup = (nk - kg < KGROUP) ? (nk - kg) : KGROUP;

        // ---- cooperative stage of this k-group's fragments into LDS ----
        {
            const uint4* gsrc = (const uint4*)(frag + (size_t)kg * 4096);
            uint4* ldst = (uint4*)sfrag;
            int nchunk = ngroup * 512;  // 16B chunks
            for (int idx = threadIdx.x; idx < nchunk; idx += 256)
                ldst[idx] = gsrc[idx];
        }
        __syncthreads();

        // ---- prefetch next k-group into cache while we compute ----
        if (kg + KGROUP < nk) {
            int nnext = ((nk - kg - KGROUP) < KGROUP) ? (nk - kg - KGROUP) : KGROUP;
            const char* nxt = (const char*)(frag + (size_t)(kg + KGROUP) * 4096);
            int nbytes = nnext * 8192;
            for (int pb = threadIdx.x * 256; pb < nbytes; pb += 256 * 256)
                __builtin_prefetch(nxt + pb, 0, 1);
        }

        if (active) {
            for (int ksl = 0; ksl < ngroup; ++ksl) {
                int ks = kg + ksl;
                int kbase = ks * 32 + khalf * 16;
                const float* arow;
                int kb;
                if (concat3) {
                    int seg = (ks * 32) >> 7;
                    kb = kbase & 127;
                    arow = (seg == 0) ? p0 : ((seg == 1) ? p1 : p2);
                } else {
                    arow = p0;
                    kb = kbase;
                }

                v16h a;
                if (concat3 || kvec) {
                    float4 f0 = *(const float4*)(arow + kb);
                    float4 f1 = *(const float4*)(arow + kb + 4);
                    float4 f2 = *(const float4*)(arow + kb + 8);
                    float4 f3 = *(const float4*)(arow + kb + 12);
                    a[0]  = rv ? (_Float16)f0.x : hz;  a[1]  = rv ? (_Float16)f0.y : hz;
                    a[2]  = rv ? (_Float16)f0.z : hz;  a[3]  = rv ? (_Float16)f0.w : hz;
                    a[4]  = rv ? (_Float16)f1.x : hz;  a[5]  = rv ? (_Float16)f1.y : hz;
                    a[6]  = rv ? (_Float16)f1.z : hz;  a[7]  = rv ? (_Float16)f1.w : hz;
                    a[8]  = rv ? (_Float16)f2.x : hz;  a[9]  = rv ? (_Float16)f2.y : hz;
                    a[10] = rv ? (_Float16)f2.z : hz;  a[11] = rv ? (_Float16)f2.w : hz;
                    a[12] = rv ? (_Float16)f3.x : hz;  a[13] = rv ? (_Float16)f3.y : hz;
                    a[14] = rv ? (_Float16)f3.z : hz;  a[15] = rv ? (_Float16)f3.w : hz;
                } else {
#pragma unroll
                    for (int e = 0; e < 8; ++e) {
                        int k = kb + e;
                        a[e] = (rv && k < K) ? (_Float16)arow[k] : hz;
                    }
#pragma unroll
                    for (int e = 0; e < 8; ++e) {
                        int k = kb + 8 + e;
                        a[8 + e] = (rv && k < K) ? (_Float16)arow[k] : hz;
                    }
                }

                const _Float16* fb = sfrag + ((size_t)ksl * 256 + lane) * 16;
#pragma unroll
                for (int nt = 0; nt < 8; ++nt) {
                    v16h b = *(const v16h*)(fb + (size_t)nt * 512);
                    acc[nt] = __builtin_amdgcn_wmma_f32_16x16x32_f16(
                        false, a, false, b, (short)0, acc[nt], false, false);
                }
            }
        }
        __syncthreads();
    }

    if (!active) return;

    // Epilogue. C layout: VGPR v -> row = row0 + v + khalf*8, col = nt*16 + (lane&15)
#pragma unroll
    for (int nt = 0; nt < 8; ++nt) {
        int col = nt * 16 + mlane;
        float bval = bias ? bias[col] : 0.0f;
#pragma unroll
        for (int v = 0; v < 8; ++v) {
            int row = row0 + (khalf << 3) + v;
            if (row >= M) continue;
            float val = acc[nt][v] + bval;
            if (act == ACT_SILU) val = silu_f(val);
            float* cp = C + (size_t)row * DIMF + col;
            if (epi == EPI_STORE)      *cp = val;
            else if (epi == EPI_ADD)   *cp += val;
            else                       *cp *= val;
        }
    }
}

// ---------------------------------------------------------------------------
// Elementwise / gather / scatter kernels
// ---------------------------------------------------------------------------
__global__ void pamnet_embed(const float* __restrict__ emb,
                             const int* __restrict__ x_raw,
                             float* __restrict__ h0, int N) {
    int id = blockIdx.x * blockDim.x + threadIdx.x;
    if (id >= N * DIMF) return;
    int v = id >> 7, c = id & 127;
    h0[id] = emb[(size_t)x_raw[v] * DIMF + c];
}

__global__ void pamnet_copy(const float* __restrict__ s, float* __restrict__ d, size_t n) {
    size_t id = (size_t)blockIdx.x * blockDim.x + threadIdx.x;
    if (id < n) d[id] = s[id];
}

__global__ void pamnet_add(const float* __restrict__ a, const float* __restrict__ b,
                           float* __restrict__ d, size_t n) {
    size_t id = (size_t)blockIdx.x * blockDim.x + threadIdx.x;
    if (id < n) d[id] = a[id] + b[id];
}

__global__ void pamnet_zero(float* __restrict__ d, size_t n) {
    size_t id = (size_t)blockIdx.x * blockDim.x + threadIdx.x;
    if (id < n) d[id] = 0.0f;
}

__global__ void pamnet_scatter_add(const float* __restrict__ src,
                                   const int* __restrict__ seg,
                                   int rows, float* __restrict__ dst) {
    int id = blockIdx.x * blockDim.x + threadIdx.x;
    if (id >= rows * DIMF) return;
    int r = id >> 7, c = id & 127;
    atomicAdd(dst + (size_t)seg[r] * DIMF + c, src[(size_t)r * DIMF + c]);
}

// dst[seg[t]] += msrc[gidx[t]] * sbf[t]
__global__ void pamnet_scatter_gmul(const float* __restrict__ msrc,
                                    const int* __restrict__ gidx,
                                    const float* __restrict__ sbf,
                                    const int* __restrict__ seg,
                                    int T, float* __restrict__ dst) {
    int id = blockIdx.x * blockDim.x + threadIdx.x;
    if (id >= T * DIMF) return;
    int t = id >> 7, c = id & 127;
    atomicAdd(dst + (size_t)seg[t] * DIMF + c,
              msrc[(size_t)gidx[t] * DIMF + c] * sbf[(size_t)t * DIMF + c]);
}

// Distance + Bessel RBF (raw, [E x 16])
__global__ void pamnet_bessel(const float* __restrict__ pos,
                              const int* __restrict__ ei, int E, float cutoff,
                              float* __restrict__ rbf_raw, float* __restrict__ dist) {
    int e = blockIdx.x * blockDim.x + threadIdx.x;
    if (e >= E) return;
    int j = ei[e], i = ei[E + e];
    float dx = pos[3 * i + 0] - pos[3 * j + 0];
    float dy = pos[3 * i + 1] - pos[3 * j + 1];
    float dz = pos[3 * i + 2] - pos[3 * j + 2];
    float d = sqrtf(dx * dx + dy * dy + dz * dz);
    dist[e] = d;
    float x = d / cutoff;
    float env = envelope_f(x);
#pragma unroll
    for (int k = 0; k < 16; ++k)
        rbf_raw[(size_t)e * 16 + k] = env * __sinf((k + 1) * PI_F * x);
}

// norms[l*6+n] = 1/sqrt(0.5 * j_{l+1}(z)^2)
__global__ void pamnet_norms(float* __restrict__ norms) {
    int id = blockIdx.x * blockDim.x + threadIdx.x;
    if (id >= NSPH * NRAD) return;
    int l = id / NRAD;
    float z = SPH_Z[id];
    float jv = sph_jn_f(z, l + 1);
    norms[id] = rsqrtf(0.5f * jv * jv);
}

// Raw SBF basis, [T x 42]: env(d)*norm[l][n]*j_l(d*z[l][n])*Y_l(cos angle)
__global__ void pamnet_sbf(const float* __restrict__ pos,
                           const float* __restrict__ dist_l,
                           const float* __restrict__ norms,
                           const int* __restrict__ na, const int* __restrict__ nb,
                           const int* __restrict__ nc,
                           const int* __restrict__ idx_edge,
                           int T, float* __restrict__ out) {
    int t = blockIdx.x * blockDim.x + threadIdx.x;
    if (t >= T) return;
    int ia = na[t], ib = nb[t], ic = nc[t];
    float v1x = pos[3 * ib + 0] - pos[3 * ia + 0];
    float v1y = pos[3 * ib + 1] - pos[3 * ia + 1];
    float v1z = pos[3 * ib + 2] - pos[3 * ia + 2];
    float v2x = pos[3 * ic + 0] - pos[3 * ib + 0];
    float v2y = pos[3 * ic + 1] - pos[3 * ib + 1];
    float v2z = pos[3 * ic + 2] - pos[3 * ib + 2];
    float dot = v1x * v2x + v1y * v2y + v1z * v2z;
    float n1 = sqrtf(v1x * v1x + v1y * v1y + v1z * v1z);
    float n2 = sqrtf(v2x * v2x + v2y * v2y + v2z * v2z);
    float c = dot / fmaxf(n1 * n2, 1e-12f);
    c = fminf(1.0f, fmaxf(-1.0f, c));

    float d = dist_l[idx_edge[t]] / CUT_L;
    float env = envelope_f(d);

    float P[NSPH];
    P[0] = 1.0f; P[1] = c;
#pragma unroll
    for (int l = 1; l < NSPH - 1; ++l)
        P[l + 1] = ((2.0f * l + 1.0f) * c * P[l] - l * P[l - 1]) / (l + 1.0f);

    float* o = out + (size_t)t * (NSPH * NRAD);
    for (int l = 0; l < NSPH; ++l) {
        float Y = sqrtf((2.0f * l + 1.0f) / (4.0f * PI_F)) * P[l];
#pragma unroll
        for (int n = 0; n < NRAD; ++n) {
            float jv = sph_jn_f(d * SPH_Z[l * NRAD + n], l);
            o[l * NRAD + n] = env * norms[l * NRAD + n] * jv * Y;
        }
    }
}

// Readout: out[g] = sum_c gsum[g][c]*head_W[c] + head_b
__global__ void pamnet_head(const float* __restrict__ gsum,
                            const float* __restrict__ hw,
                            const float* __restrict__ hb,
                            float* __restrict__ out) {
    __shared__ float red[DIMF];
    int g = blockIdx.x, c = threadIdx.x;
    red[c] = gsum[(size_t)g * DIMF + c] * hw[c];
    __syncthreads();
    for (int s = 64; s > 0; s >>= 1) {
        if (c < s) red[c] += red[c + s];
        __syncthreads();
    }
    if (c == 0) out[g] = red[0] + hb[0];
}

// ---------------------------------------------------------------------------
// Host-side helpers
// ---------------------------------------------------------------------------
static void run_gemm_plain(hipStream_t s, const float* A, const float* W, const float* b,
                           float* C, int M, int K, int act, int epi, _Float16* frag) {
    if (M <= 0) return;
    int nk = (K + 31) / 32, tot = nk * 4096;
    pamnet_pack_w<<<(tot + 255) / 256, 256, 0, s>>>(W, K, frag, tot);
    int tiles = (M + 15) / 16;
    pamnet_gemm<<<(tiles + GEMM_WAVES - 1) / GEMM_WAVES, 32 * GEMM_WAVES, 0, s>>>(
        A, K, K, nullptr, nullptr, nullptr, nullptr, nullptr, 0,
        frag, b, C, M, act, epi);
}

static void run_gemm_concat(hipStream_t s, const float* GA, const float* GB, const float* GC,
                            const int* idxA, const int* idxB,
                            const float* W, const float* b,
                            float* C, int M, _Float16* frag) {
    if (M <= 0) return;
    int K = 3 * DIMF, nk = (K + 31) / 32, tot = nk * 4096;
    pamnet_pack_w<<<(tot + 255) / 256, 256, 0, s>>>(W, K, frag, tot);
    int tiles = (M + 15) / 16;
    pamnet_gemm<<<(tiles + GEMM_WAVES - 1) / GEMM_WAVES, 32 * GEMM_WAVES, 0, s>>>(
        nullptr, K, 0, GA, GB, GC, idxA, idxB, 1,
        frag, b, C, M, ACT_SILU, EPI_STORE);
}

static inline size_t align256(size_t x) { return (x + 255) & ~(size_t)255; }

// Params pytree leaf indexing (jax order: dict keys sorted, lists in order)
#define I_POS 0
#define I_EMB 1
#define I_FEAT_W 2
#define I_FEAT_B 3
#define GBASE 4
#define GSTRIDE 21
#define I_HEADW 67
#define I_HEADB 68
#define LBASE 69
#define LSTRIDE 26
#define I_RBFG_W 147
#define I_RBFG_B 148
#define I_RBFL_W 149
#define I_RBFL_B 150
#define I_SBF1_W 151
#define I_SBF1_B 152
#define I_SBF2_W 153
#define I_SBF2_B 154
#define I_XRAW 155
#define I_BATCH 156
#define I_EIL 157
#define I_EIG 158
#define I_IDXI 159
#define I_IDXJ 160
#define I_IDXK 161
#define I_IDXKJ 162
#define I_IDXJI 163
#define I_IP 164
#define I_J1P 165
#define I_J2P 166
#define I_JJP 167
#define I_JIP 168

extern "C" void kernel_launch(void* const* d_in, const int* in_sizes, int n_in,
                              void* d_out, int out_size, void* d_ws, size_t ws_size,
                              hipStream_t stream) {
    auto F = [&](int i) -> const float* { return (const float*)d_in[i]; };
    auto I = [&](int i) -> const int*   { return (const int*)d_in[i]; };

    const int N  = in_sizes[I_XRAW];
    const int El = in_sizes[I_EIL] / 2;
    const int Eg = in_sizes[I_EIG] / 2;
    const int T  = in_sizes[I_IDXKJ];
    const int P  = in_sizes[I_JJP];
    const int G  = out_size;
    const int Emax = (Eg > El) ? Eg : El;

    const float* pos = F(I_POS);
    const int* eil = I(I_EIL);
    const int* eig = I(I_EIG);
    const int* jl = eil;            const int* il = eil + El;
    const int* jg = eig;            const int* ig = eig + Eg;

    // ---- workspace bump allocation ----
    char* wsb = (char*)d_ws;
    size_t off = 0;
    auto alloc = [&](size_t bytes) -> void* {
        void* p = wsb + off;
        off = align256(off + bytes);
        return p;
    };
    _Float16* frag = (_Float16*)alloc((size_t)384 * DIMF * sizeof(_Float16));
    float* norms   = (float*)alloc(NSPH * NRAD * sizeof(float));
    float* h   = (float*)alloc((size_t)N * DIMF * 4);
    float* t0  = (float*)alloc((size_t)N * DIMF * 4);
    float* t1  = (float*)alloc((size_t)N * DIMF * 4);
    float* t2  = (float*)alloc((size_t)N * DIMF * 4);
    float* dist_l   = (float*)alloc((size_t)El * 4);
    float* dist_g   = (float*)alloc((size_t)Eg * 4);
    float* rbfraw_l = (float*)alloc((size_t)El * 16 * 4);
    float* rbfraw_g = (float*)alloc((size_t)Eg * 16 * 4);
    float* rbf_l = (float*)alloc((size_t)El * DIMF * 4);
    float* rbf_g = (float*)alloc((size_t)Eg * DIMF * 4);
    float* sbfraw2 = (float*)alloc((size_t)T * 42 * 4);
    float* sbfraw1 = (float*)alloc((size_t)P * 42 * 4);
    float* sbf2h = (float*)alloc((size_t)T * DIMF * 4);
    float* sbf1h = (float*)alloc((size_t)P * DIMF * 4);
    float* mkj = (float*)alloc((size_t)El * DIMF * 4);
    float* mjj = (float*)alloc((size_t)El * DIMF * 4);
    float* m_e = (float*)alloc((size_t)Emax * DIMF * 4);
    float* gsum = (float*)alloc((size_t)G * DIMF * 4);
    (void)ws_size; (void)n_in;

    auto gridE = [](long long n) { return dim3((unsigned)((n + 255) / 256)); };

    // ---- preprocessing ----
    pamnet_norms<<<1, 64, 0, stream>>>(norms);

    pamnet_embed<<<gridE((long long)N * DIMF), 256, 0, stream>>>(F(I_EMB), I(I_XRAW), t0, N);
    run_gemm_plain(stream, t0, F(I_FEAT_W), F(I_FEAT_B), h, N, DIMF, ACT_SILU, EPI_STORE, frag);

    if (Eg > 0) pamnet_bessel<<<gridE(Eg), 256, 0, stream>>>(pos, eig, Eg, CUT_G, rbfraw_g, dist_g);
    if (El > 0) pamnet_bessel<<<gridE(El), 256, 0, stream>>>(pos, eil, El, CUT_L, rbfraw_l, dist_l);

    run_gemm_plain(stream, rbfraw_g, F(I_RBFG_W), F(I_RBFG_B), rbf_g, Eg, 16, ACT_SILU, EPI_STORE, frag);
    run_gemm_plain(stream, rbfraw_l, F(I_RBFL_W), F(I_RBFL_B), rbf_l, El, 16, ACT_SILU, EPI_STORE, frag);

    if (T > 0)
        pamnet_sbf<<<gridE(T), 256, 0, stream>>>(pos, dist_l, norms,
            I(I_IDXI), I(I_IDXJ), I(I_IDXK), I(I_IDXKJ), T, sbfraw2);
    if (P > 0)
        pamnet_sbf<<<gridE(P), 256, 0, stream>>>(pos, dist_l, norms,
            I(I_IP), I(I_J1P), I(I_J2P), I(I_JJP), P, sbfraw1);

    run_gemm_plain(stream, sbfraw2, F(I_SBF2_W), F(I_SBF2_B), sbf2h, T, 42, ACT_SILU, EPI_STORE, frag);
    run_gemm_plain(stream, sbfraw1, F(I_SBF1_W), F(I_SBF1_B), sbf1h, P, 42, ACT_SILU, EPI_STORE, frag);

    const size_t nodeElems = (size_t)N * DIMF;

    // ---- layers ----
    for (int l = 0; l < 3; ++l) {
        // ================= global MP =================
        int gb = GBASE + GSTRIDE * l;
        // x = mlp(x1, h)
        run_gemm_plain(stream, h, F(gb + 17), F(gb + 18), t0, N, DIMF, ACT_SILU, EPI_STORE, frag);
        // m = mlp(m, concat(x[i], x[j], rbf_g))
        run_gemm_concat(stream, t0, t0, rbf_g, ig, jg, F(gb + 1), F(gb + 2), m_e, Eg, frag);
        // x = segment_sum(m, i) + h
        pamnet_copy<<<gridE(nodeElems), 256, 0, stream>>>(h, t1, nodeElems);
        if (Eg > 0)
            pamnet_scatter_add<<<gridE((long long)Eg * DIMF), 256, 0, stream>>>(m_e, ig, Eg, t1);
        // t0 = mlp(x2, x); t0 = res_apply(res1, t0); h = t0 + t1
        run_gemm_plain(stream, t1, F(gb + 19), F(gb + 20), t0, N, DIMF, ACT_SILU, EPI_STORE, frag);
        run_gemm_plain(stream, t0, F(gb + 5),  F(gb + 6),  t2, N, DIMF, ACT_SILU, EPI_STORE, frag);
        run_gemm_plain(stream, t2, F(gb + 7),  F(gb + 8),  t0, N, DIMF, ACT_SILU, EPI_ADD,   frag);
        pamnet_add<<<gridE(nodeElems), 256, 0, stream>>>(t0, t1, h, nodeElems);
        // h = res_apply(res3, res_apply(res2, h))
        run_gemm_plain(stream, h,  F(gb + 9),  F(gb + 10), t2, N, DIMF, ACT_SILU, EPI_STORE, frag);
        run_gemm_plain(stream, t2, F(gb + 11), F(gb + 12), h,  N, DIMF, ACT_SILU, EPI_ADD,   frag);
        run_gemm_plain(stream, h,  F(gb + 13), F(gb + 14), t2, N, DIMF, ACT_SILU, EPI_STORE, frag);
        run_gemm_plain(stream, t2, F(gb + 15), F(gb + 16), h,  N, DIMF, ACT_SILU, EPI_ADD,   frag);

        // ================= local MP =================
        int lb = LBASE + LSTRIDE * l;
        // x = mlp(x1, h)
        run_gemm_plain(stream, h, F(lb + 22), F(lb + 23), t0, N, DIMF, ACT_SILU, EPI_STORE, frag);
        // feat-based messages
        run_gemm_concat(stream, t0, t0, rbf_l, il, jl, F(lb + 5), F(lb + 6), mkj, El, frag); // kj
        run_gemm_concat(stream, t0, t0, rbf_l, il, jl, F(lb + 3), F(lb + 4), mjj, El, frag); // ji2
        run_gemm_concat(stream, t0, t0, rbf_l, il, jl, F(lb + 1), F(lb + 2), m_e, El, frag); // ji1
        // m += segment_sum(m_kj[idx_kj]*sbf2, idx_ji); m += segment_sum(m_jj[idx_jj]*sbf1, idx_ji_pair)
        if (T > 0)
            pamnet_scatter_gmul<<<gridE((long long)T * DIMF), 256, 0, stream>>>(
                mkj, I(I_IDXKJ), sbf2h, I(I_IDXJI), T, m_e);
        if (P > 0)
            pamnet_scatter_gmul<<<gridE((long long)P * DIMF), 256, 0, stream>>>(
                mjj, I(I_JJP), sbf1h, I(I_JIP), P, m_e);
        // m *= (rbf_l @ rbf_gate)
        run_gemm_plain(stream, rbf_l, F(lb + 21), nullptr, m_e, El, DIMF, ACT_NONE, EPI_MUL, frag);
        // x = segment_sum(m, i) + h
        pamnet_copy<<<gridE(nodeElems), 256, 0, stream>>>(h, t1, nodeElems);
        if (El > 0)
            pamnet_scatter_add<<<gridE((long long)El * DIMF), 256, 0, stream>>>(m_e, il, El, t1);
        // x2 + residual chain
        run_gemm_plain(stream, t1, F(lb + 24), F(lb + 25), t0, N, DIMF, ACT_SILU, EPI_STORE, frag);
        run_gemm_plain(stream, t0, F(lb + 9),  F(lb + 10), t2, N, DIMF, ACT_SILU, EPI_STORE, frag);
        run_gemm_plain(stream, t2, F(lb + 11), F(lb + 12), t0, N, DIMF, ACT_SILU, EPI_ADD,   frag);
        pamnet_add<<<gridE(nodeElems), 256, 0, stream>>>(t0, t1, h, nodeElems);
        run_gemm_plain(stream, h,  F(lb + 13), F(lb + 14), t2, N, DIMF, ACT_SILU, EPI_STORE, frag);
        run_gemm_plain(stream, t2, F(lb + 15), F(lb + 16), h,  N, DIMF, ACT_SILU, EPI_ADD,   frag);
        run_gemm_plain(stream, h,  F(lb + 17), F(lb + 18), t2, N, DIMF, ACT_SILU, EPI_STORE, frag);
        run_gemm_plain(stream, t2, F(lb + 19), F(lb + 20), h,  N, DIMF, ACT_SILU, EPI_ADD,   frag);
    }

    // ---- readout ----
    pamnet_zero<<<gridE((long long)G * DIMF), 256, 0, stream>>>(gsum, (size_t)G * DIMF);
    pamnet_scatter_add<<<gridE((long long)N * DIMF), 256, 0, stream>>>(h, I(I_BATCH), N, gsum);
    pamnet_head<<<dim3(G), dim3(DIMF), 0, stream>>>(gsum, F(I_HEADW), F(I_HEADB), (float*)d_out);
}
